// VARGR_AGREE2_20091857010786
// MI455X (gfx1250) — compile-verified, matrix operand order checked by reading, not run on hardware
//
#include <hip/hip_runtime.h>
#include <hip/hip_bf16.h>
#include <math.h>

typedef __attribute__((ext_vector_type(4)))  float    v4f;
typedef __attribute__((ext_vector_type(4)))  _Float16 v4h;
typedef __attribute__((ext_vector_type(8)))  _Float16 v8h;
typedef __attribute__((ext_vector_type(16))) _Float16 v16h;
typedef __attribute__((ext_vector_type(8)))  float    v8f;

#define GSZ 20
#define DIM 64
#define HGE 96

static __device__ __forceinline__ v8f zero8() {
  v8f z = {0.f,0.f,0.f,0.f,0.f,0.f,0.f,0.f};
  return z;
}

static __device__ __forceinline__ v4h cvt4(v4f x) {
  v4h h = { (_Float16)x.x, (_Float16)x.y, (_Float16)x.z, (_Float16)x.w };
  return h;
}

// Load a 16x32 f16 fragment (A-layout; B uses the same layout on a transposed
// weight tile) from an LDS tile with row stride `stride` halfs.
// Lane L<16: row L, K = {0..7, 16..23}; lane>=16: row L-16, K = {8..15, 24..31}.
static __device__ __forceinline__ v16h load_frag(const _Float16* base, int stride) {
  int lane = threadIdx.x & 31;
  int r  = lane & 15;
  int hi = lane >> 4;
  const _Float16* p = base + r * stride + hi * 8;
  v8h lo = *(const v8h*)(p);
  v8h hh = *(const v8h*)(p + 16);
  return __builtin_shufflevector(lo, hh, 0,1,2,3,4,5,6,7,8,9,10,11,12,13,14,15);
}

static __device__ __forceinline__ v8f wmma16(v16h a, v16h b, v8f c) {
  return __builtin_amdgcn_wmma_f32_16x16x32_f16(false, a, false, b, (short)0, c, false, false);
}

__global__ void k_zero(float* __restrict__ p, int n) {
  int i = blockIdx.x * blockDim.x + threadIdx.x;
  if (i < n) p[i] = 0.f;
}

// ---------------------------------------------------------------------------
// Kernel 1: per-member encoder MLP (64 -> 64 -> 64) + mean over group.
// 8 waves/block, each wave owns 16 rows of the flattened (B*G, 64) matrix.
// ---------------------------------------------------------------------------
__global__ __launch_bounds__(256) void k_member(
    const int*   __restrict__ gin,  const int* __restrict__ gmem,
    const float* __restrict__ uemb,
    const float* __restrict__ w1, const float* __restrict__ b1,
    const float* __restrict__ w2, const float* __restrict__ b2,
    float* __restrict__ gzacc)
{
  __shared__ alignas(16) _Float16 w1t[DIM * DIM];      // [n][k] transposed
  __shared__ alignas(16) _Float16 w2t[DIM * DIM];
  __shared__ alignas(16) _Float16 meT[8][16 * DIM];
  __shared__ alignas(16) _Float16 hid[8][16 * DIM];

  int tid = threadIdx.x;
  for (int i = tid; i < (DIM * DIM) / 4; i += 256) {   // w: (64,64) flat4
    int k = i >> 4;                                    // i*4 = k*64 + n
    int n = (i & 15) << 2;
    v4f a = *(const v4f*)(w1 + (i << 2));
    v4f c = *(const v4f*)(w2 + (i << 2));
    w1t[(n + 0) * DIM + k] = (_Float16)a.x;
    w1t[(n + 1) * DIM + k] = (_Float16)a.y;
    w1t[(n + 2) * DIM + k] = (_Float16)a.z;
    w1t[(n + 3) * DIM + k] = (_Float16)a.w;
    w2t[(n + 0) * DIM + k] = (_Float16)c.x;
    w2t[(n + 1) * DIM + k] = (_Float16)c.y;
    w2t[(n + 2) * DIM + k] = (_Float16)c.z;
    w2t[(n + 3) * DIM + k] = (_Float16)c.w;
  }
  int wave = tid >> 5, lane = tid & 31;
  int row0 = (blockIdx.x * 8 + wave) * 16;
  for (int i = lane; i < 16 * (DIM / 4); i += 32) {    // gather member rows
    int r  = i >> 4;
    int c4 = (i & 15) << 2;
    int grow = row0 + r;
    int b = grow / GSZ;
    int g = grow - b * GSZ;
    int uidx = gmem[gin[b] * GSZ + g];
    v4f x = *(const v4f*)(uemb + (size_t)uidx * DIM + c4);
    *(v4h*)(&meT[wave][r * DIM + c4]) = cvt4(x);
  }
  __syncthreads();

  int nlo = lane & 15, hh = lane >> 4;

  v8f acc[4];
  #pragma unroll
  for (int n = 0; n < 4; n++) acc[n] = zero8();
  #pragma unroll
  for (int k0 = 0; k0 < DIM; k0 += 32) {
    v16h a = load_frag(&meT[wave][k0], DIM);
    #pragma unroll
    for (int n = 0; n < 4; n++) {
      v16h bf = load_frag(&w1t[n * 16 * DIM + k0], DIM);
      acc[n] = wmma16(a, bf, acc[n]);
    }
  }
  #pragma unroll
  for (int n = 0; n < 4; n++) {
    float bias = b1[n * 16 + nlo];
    #pragma unroll
    for (int v = 0; v < 8; v++) {
      float x = acc[n][v] + bias;
      hid[wave][(hh * 8 + v) * DIM + n * 16 + nlo] = (_Float16)(x > 0.f ? x : 0.f);
    }
  }
  __syncthreads();

  #pragma unroll
  for (int n = 0; n < 4; n++) acc[n] = zero8();
  #pragma unroll
  for (int k0 = 0; k0 < DIM; k0 += 32) {
    v16h a = load_frag(&hid[wave][k0], DIM);
    #pragma unroll
    for (int n = 0; n < 4; n++) {
      v16h bf = load_frag(&w2t[n * 16 * DIM + k0], DIM);
      acc[n] = wmma16(a, bf, acc[n]);
    }
  }
  // Mean over group: the 8 rows a lane holds are consecutive global rows, so
  // they span at most one group boundary -> flush <=2 atomics per n-tile.
  #pragma unroll
  for (int n = 0; n < 4; n++) {
    float bias = b2[n * 16 + nlo];
    int   col  = n * 16 + nlo;
    float part = 0.f;
    int   curb = (row0 + hh * 8) / GSZ;
    #pragma unroll
    for (int v = 0; v < 8; v++) {
      int grow = row0 + hh * 8 + v;
      int b = grow / GSZ;
      if (b != curb) {
        atomicAdd(&gzacc[curb * DIM + col], part * (1.0f / GSZ));
        part = 0.f;
        curb = b;
      }
      part += acc[n][v] + bias;
    }
    atomicAdd(&gzacc[curb * DIM + col], part * (1.0f / GSZ));
  }
}

// ---------------------------------------------------------------------------
// Kernel 2: group encoder relu(mean) -> 96 -> 128, then mu + sigma*eps.
// 4 waves/block (LDS < 64KB), 16 batch rows per wave.
// ---------------------------------------------------------------------------
__global__ __launch_bounds__(128) void k_group(
    const float* __restrict__ gzacc, const float* __restrict__ eps,
    const float* __restrict__ w1, const float* __restrict__ b1,
    const float* __restrict__ w2, const float* __restrict__ b2,
    float* __restrict__ muSig)
{
  __shared__ alignas(16) _Float16 w1t[HGE * DIM];      // [n<96][k<64]
  __shared__ alignas(16) _Float16 w2t[128 * HGE];      // [n<128][k<96]
  __shared__ alignas(16) _Float16 inT[4][16 * DIM];
  __shared__ alignas(16) _Float16 hid[4][16 * HGE];

  int tid = threadIdx.x;
  for (int i = tid; i < (DIM * HGE) / 4; i += 128) {   // w1: (64,96), 96%4==0
    int k = i / 24;                                    // i*4 = k*96 + n
    int n = (i - k * 24) << 2;
    v4f a = *(const v4f*)(w1 + (i << 2));
    w1t[(n + 0) * DIM + k] = (_Float16)a.x;
    w1t[(n + 1) * DIM + k] = (_Float16)a.y;
    w1t[(n + 2) * DIM + k] = (_Float16)a.z;
    w1t[(n + 3) * DIM + k] = (_Float16)a.w;
  }
  for (int i = tid; i < (HGE * 128) / 4; i += 128) {   // w2: (96,128)
    int k = i >> 5;                                    // i*4 = k*128 + n
    int n = (i & 31) << 2;
    v4f a = *(const v4f*)(w2 + (i << 2));
    w2t[(n + 0) * HGE + k] = (_Float16)a.x;
    w2t[(n + 1) * HGE + k] = (_Float16)a.y;
    w2t[(n + 2) * HGE + k] = (_Float16)a.z;
    w2t[(n + 3) * HGE + k] = (_Float16)a.w;
  }
  int wave = tid >> 5, lane = tid & 31;
  int row0 = (blockIdx.x * 4 + wave) * 16;
  for (int i = lane; i < (16 * DIM) / 4; i += 32) {    // gz = relu(mean)
    v4f x = *(const v4f*)(gzacc + row0 * DIM + (i << 2));
    v4f r = { x.x > 0.f ? x.x : 0.f, x.y > 0.f ? x.y : 0.f,
              x.z > 0.f ? x.z : 0.f, x.w > 0.f ? x.w : 0.f };
    *(v4h*)(&inT[wave][i << 2]) = cvt4(r);
  }
  __syncthreads();

  int nlo = lane & 15, hh = lane >> 4;

  v8f acc[6];
  #pragma unroll
  for (int n = 0; n < 6; n++) acc[n] = zero8();
  #pragma unroll
  for (int k0 = 0; k0 < DIM; k0 += 32) {
    v16h a = load_frag(&inT[wave][k0], DIM);
    #pragma unroll
    for (int n = 0; n < 6; n++) {
      v16h bf = load_frag(&w1t[n * 16 * DIM + k0], DIM);
      acc[n] = wmma16(a, bf, acc[n]);
    }
  }
  #pragma unroll
  for (int n = 0; n < 6; n++) {
    float bias = b1[n * 16 + nlo];
    #pragma unroll
    for (int v = 0; v < 8; v++) {
      float x = acc[n][v] + bias;
      hid[wave][(hh * 8 + v) * HGE + n * 16 + nlo] = (_Float16)(x > 0.f ? x : 0.f);
    }
  }
  __syncthreads();

  v8f a2[8];
  #pragma unroll
  for (int n = 0; n < 8; n++) a2[n] = zero8();
  #pragma unroll
  for (int k0 = 0; k0 < HGE; k0 += 32) {
    v16h a = load_frag(&hid[wave][k0], HGE);
    #pragma unroll
    for (int n = 0; n < 8; n++) {
      v16h bf = load_frag(&w2t[n * 16 * HGE + k0], HGE);
      a2[n] = wmma16(a, bf, a2[n]);
    }
  }
  // column d (mu, tile n) and column 64+d (log_sigma, tile n+4) live in the
  // same lane & same accumulator slot -> reparameterize fully in registers.
  #pragma unroll
  for (int n = 0; n < 4; n++) {
    float bmu = b2[n * 16 + nlo];
    float bls = b2[(n + 4) * 16 + nlo];
    #pragma unroll
    for (int v = 0; v < 8; v++) {
      float mu = a2[n][v] + bmu;
      float ls = a2[n + 4][v] + bls;
      float sg = 0.1f + 0.9f / (1.f + expf(-ls));
      int grow = row0 + hh * 8 + v;
      int d = n * 16 + nlo;
      muSig[grow * DIM + d] = mu + sg * eps[grow * DIM + d];
    }
  }
}

// ---------------------------------------------------------------------------
// Kernel 3: attention over members (WMMA for the 128->16 layer), softmax,
// weighted sum, reparam add, NCF predictor, sigmoid.  1 wave per batch elem.
// ---------------------------------------------------------------------------
__global__ __launch_bounds__(128) void k_attn(
    const int*   __restrict__ gin,  const int* __restrict__ iin,
    const int*   __restrict__ gmem,
    const float* __restrict__ uemb, const float* __restrict__ iemb,
    const float* __restrict__ aw1, const float* __restrict__ ab1,
    const float* __restrict__ aw2, const float* __restrict__ ab2,
    const float* __restrict__ pw1, const float* __restrict__ pb1,
    const float* __restrict__ pw2, const float* __restrict__ pb2,
    const float* __restrict__ muSig, float* __restrict__ out)
{
  __shared__ alignas(16) _Float16 awt[16 * 128];       // [n<16][k<128]
  __shared__ alignas(16) _Float16 gi[4][32 * 128];     // [member row][me|ie]
  __shared__ float h1[4][32 * 16];
  __shared__ float sc[4][32];
  __shared__ float wv[4][32];
  __shared__ float gef[4][DIM];
  __shared__ float iev[4][DIM];

  int tid = threadIdx.x;
  for (int i = tid; i < (128 * 16) / 4; i += 128) {    // at_w1: (128,16)
    int k = i >> 2;                                    // i*4 = k*16 + n
    int n = (i & 3) << 2;
    v4f a = *(const v4f*)(aw1 + (i << 2));
    awt[(n + 0) * 128 + k] = (_Float16)a.x;
    awt[(n + 1) * 128 + k] = (_Float16)a.y;
    awt[(n + 2) * 128 + k] = (_Float16)a.z;
    awt[(n + 3) * 128 + k] = (_Float16)a.w;
  }
  int wave = tid >> 5, lane = tid & 31;
  int b    = blockIdx.x * 4 + wave;
  int gid  = gin[b];
  int item = iin[b];

  for (int i = lane; i < 32 * 32; i += 32) {           // 32 rows x 32 float4s
    int r = i >> 5;
    int q = (i & 31) << 2;                             // col 0,4,...,124
    v4h h = { (_Float16)0.f, (_Float16)0.f, (_Float16)0.f, (_Float16)0.f };
    if (r < GSZ) {
      v4f x;
      if (q < DIM) {
        int uidx = gmem[gid * GSZ + r];
        x = *(const v4f*)(uemb + (size_t)uidx * DIM + q);
      } else {
        x = *(const v4f*)(iemb + (size_t)item * DIM + (q - DIM));
      }
      h = cvt4(x);
    }
    *(v4h*)(&gi[wave][r * 128 + q]) = h;
  }
  __syncthreads();

  int nlo = lane & 15, hh = lane >> 4;
  #pragma unroll
  for (int t = 0; t < 2; t++) {                        // two 16-row tiles
    v8f acc = zero8();
    #pragma unroll
    for (int k0 = 0; k0 < 128; k0 += 32) {
      v16h a  = load_frag(&gi[wave][t * 16 * 128 + k0], 128);
      v16h bf = load_frag(&awt[k0], 128);
      acc = wmma16(a, bf, acc);
    }
    float bias = ab1[nlo];
    #pragma unroll
    for (int v = 0; v < 8; v++) {
      float x = acc[v] + bias;
      h1[wave][(t * 16 + hh * 8 + v) * 16 + nlo] = x > 0.f ? x : 0.f;
    }
  }
  __syncthreads();

  if (lane < GSZ) {                                    // 16 -> 1 score head
    float s = ab2[0];
    for (int j = 0; j < 16; j++) s += h1[wave][lane * 16 + j] * aw2[j];
    sc[wave][lane] = s;
  }
  __syncthreads();

  float mx = -1e30f;
  for (int g = 0; g < GSZ; g++) mx = fmaxf(mx, sc[wave][g]);
  float sum = 0.f;
  for (int g = 0; g < GSZ; g++) sum += expf(sc[wave][g] - mx);
  if (lane < GSZ) wv[wave][lane] = expf(sc[wave][lane] - mx) / sum;
  __syncthreads();

  for (int d = lane; d < DIM; d += 32) {               // g_att + mu + sigma*eps
    float ga = 0.f;
    for (int g = 0; g < GSZ; g++) ga += wv[wave][g] * (float)gi[wave][g * 128 + d];
    gef[wave][d] = ga + muSig[b * DIM + d];
    iev[wave][d] = iemb[(size_t)item * DIM + d];
  }
  __syncthreads();

  if (lane < 8) {                                      // ncf 192 -> 8, relu
    float a = pb1[lane];
    for (int t = 0; t < 3 * DIM; t++) {
      float f = (t < DIM) ? gef[wave][t] * iev[wave][t]
              : (t < 2 * DIM ? gef[wave][t - DIM] : iev[wave][t - 2 * DIM]);
      a += f * pw1[t * 8 + lane];
    }
    sc[wave][lane] = a > 0.f ? a : 0.f;
  }
  __syncthreads();
  if (lane == 0) {                                     // 8 -> 1, sigmoid
    float o = pb2[0];
    for (int j = 0; j < 8; j++) o += sc[wave][j] * pw2[j];
    out[b] = 1.f / (1.f + expf(-o));
  }
}

extern "C" void kernel_launch(void* const* d_in, const int* in_sizes, int n_in,
                              void* d_out, int out_size, void* d_ws, size_t ws_size,
                              hipStream_t stream) {
  const int*   gin   = (const int*)d_in[0];
  const int*   iin   = (const int*)d_in[1];
  const int*   gmem  = (const int*)d_in[2];
  const float* eps   = (const float*)d_in[3];
  const float* uemb  = (const float*)d_in[4];
  const float* iemb  = (const float*)d_in[5];
  const float* ue_w1 = (const float*)d_in[6];
  const float* ue_b1 = (const float*)d_in[7];
  const float* ue_w2 = (const float*)d_in[8];
  const float* ue_b2 = (const float*)d_in[9];
  const float* ge_w1 = (const float*)d_in[10];
  const float* ge_b1 = (const float*)d_in[11];
  const float* ge_w2 = (const float*)d_in[12];
  const float* ge_b2 = (const float*)d_in[13];
  const float* at_w1 = (const float*)d_in[14];
  const float* at_b1 = (const float*)d_in[15];
  const float* at_w2 = (const float*)d_in[16];
  const float* at_b2 = (const float*)d_in[17];
  const float* pr_w1 = (const float*)d_in[18];
  const float* pr_b1 = (const float*)d_in[19];
  const float* pr_w2 = (const float*)d_in[20];
  const float* pr_b2 = (const float*)d_in[21];

  int B = in_sizes[0];
  float* gzacc = (float*)d_ws;                 // B*64 f32 accumulator
  float* muSig = gzacc + (size_t)B * DIM;      // B*64 f32 mu + sigma*eps
  float* outp  = (float*)d_out;

  int n = B * DIM;
  k_zero<<<(n + 255) / 256, 256, 0, stream>>>(gzacc, n);

  int rows = B * GSZ;                          // 327680, divisible by 128
  k_member<<<rows / (16 * 8), 256, 0, stream>>>(gin, gmem, uemb,
                                                ue_w1, ue_b1, ue_w2, ue_b2, gzacc);
  k_group<<<B / (16 * 4), 128, 0, stream>>>(gzacc, eps,
                                            ge_w1, ge_b1, ge_w2, ge_b2, muSig);
  k_attn<<<B / 4, 128, 0, stream>>>(gin, iin, gmem, uemb, iemb,
                                    at_w1, at_b1, at_w2, at_b2,
                                    pr_w1, pr_b1, pr_w2, pr_b2, muSig, outp);
}
